// ModelNCF_75247827026426
// MI455X (gfx1250) — compile-verified, matrix operand order checked by reading, not run on hardware
//
#include <hip/hip_runtime.h>
#include <hip/hip_bf16.h>

typedef __attribute__((ext_vector_type(16))) _Float16 v16h;
typedef __attribute__((ext_vector_type(8)))  _Float16 v8h;
typedef __attribute__((ext_vector_type(4)))  _Float16 v4h;
typedef __attribute__((ext_vector_type(8)))  float    v8f;

#define EMB 128
#define NU  256

// Load a 16x32 (rows x K) f16 WMMA fragment from row-major storage with row
// stride EMB halves. Per the CDNA5 ISA 16-bit A-matrix layout:
//   lanes 0-15 : halves 0..7 = K 0..7,  halves 8..15 = K 16..23
//   lanes 16-31: halves 0..7 = K 8..15, halves 8..15 = K 24..31
__device__ __forceinline__ v16h load_frag_g(const _Float16* base, int row,
                                            int k0, int laneHi) {
  const _Float16* p = base + (size_t)row * EMB + k0 + (laneHi ? 8 : 0);
  v8h lo = *(const v8h*)p;
  v8h hi = *(const v8h*)(p + 16);
  v16h r;
#pragma unroll
  for (int t = 0; t < 8; ++t) { r[t] = lo[t]; r[t + 8] = hi[t]; }
  return r;
}

// ---------------------------------------------------------------------------
// K1: gather user/item embedding rows (they are model outputs), plus f16
// staging copies: h16[i][e], wg16[u][e] = w[u][e]*w_gmf[e], w2_16[n][k].
// ---------------------------------------------------------------------------
__global__ void k_prep_gather(const int* __restrict__ uidx,
                              const int* __restrict__ iidx,
                              const float* __restrict__ user_emb,
                              const float* __restrict__ item_emb,
                              const float* __restrict__ W2,
                              const float* __restrict__ w_gmf,
                              float* __restrict__ w_out,
                              float* __restrict__ h_out,
                              _Float16* __restrict__ h16,
                              _Float16* __restrict__ wg16,
                              _Float16* __restrict__ w216,
                              int nI) {
  int t = blockIdx.x * blockDim.x + threadIdx.x;
  int nH = nI * EMB;
  if (t < nH) {
    int r = t >> 7, e = t & 127;
    float v = item_emb[(size_t)iidx[r] * EMB + e];
    h_out[t] = v;
    h16[t]   = (_Float16)v;
    return;
  }
  t -= nH;
  if (t < NU * EMB) {
    int r = t >> 7, e = t & 127;
    float v = user_emb[(size_t)uidx[r] * EMB + e];
    w_out[t] = v;
    wg16[t]  = (_Float16)(v * w_gmf[e]);
    return;
  }
  t -= NU * EMB;
  if (t < 64 * EMB) w216[t] = (_Float16)W2[t];
}

// ---------------------------------------------------------------------------
// K2: separable layer-1 precompute.
//   A[u][e] = b1[e] + sum_k w[u][k] * W1[e][k]        (user half of W1)
//   B[i][e] =         sum_k h[i][k] * W1[e][128 + k]  (item half of W1)
// ---------------------------------------------------------------------------
__global__ void k_prep_mlp(const float* __restrict__ w_out,
                           const float* __restrict__ h_out,
                           const float* __restrict__ W1,
                           const float* __restrict__ b1,
                           float* __restrict__ A_mlp,
                           float* __restrict__ B_mlp) {
  __shared__ float row[EMB];
  int r = blockIdx.x;
  int t = threadIdx.x;
  const float* src;
  float* dst;
  int koff;
  float bias;
  if (r < NU) {
    src = w_out + (size_t)r * EMB;  dst = A_mlp + (size_t)r * EMB;
    koff = 0;                       bias = b1[t];
  } else {
    src = h_out + (size_t)(r - NU) * EMB;  dst = B_mlp + (size_t)(r - NU) * EMB;
    koff = EMB;                            bias = 0.0f;
  }
  row[t] = src[t];
  __syncthreads();
  const float4* wrow = (const float4*)(W1 + (size_t)t * (2 * EMB) + koff);
  float acc = bias;
#pragma unroll 8
  for (int k = 0; k < EMB / 4; ++k) {
    float4 wv = wrow[k];
    acc += row[4 * k + 0] * wv.x + row[4 * k + 1] * wv.y +
           row[4 * k + 2] * wv.z + row[4 * k + 3] * wv.w;
  }
  dst[t] = acc;
}

// ---------------------------------------------------------------------------
// K3: GMF head as WMMA GEMM: pred[u,i] = sum_e wg16[u][e] * h16[i][e].
// One wave per 16x16 output tile, K=128 in 4 WMMA steps. NI_C=compile-time
// row stride (0 = runtime) so stores use immediate offsets.
// ---------------------------------------------------------------------------
template <int NI_C>
__global__ void k_gmf_wmma(const _Float16* __restrict__ wg16,
                           const _Float16* __restrict__ h16,
                           float* __restrict__ pred, int nI_rt, int nTiles) {
  const int nI = NI_C ? NI_C : nI_rt;
  int wave   = (blockIdx.x * blockDim.x + threadIdx.x) >> 5;
  if (wave >= nTiles) return;  // wave-uniform guard (EXEC stays all-ones)
  int lane   = threadIdx.x & 31;
  int laneHi = lane >> 4;
  int ln     = lane & 15;
  int tilesPerRow = nI >> 4;
  int u0 = (wave / tilesPerRow) << 4;
  int i0 = (wave % tilesPerRow) << 4;

  v8f acc = {};
#pragma unroll
  for (int kb = 0; kb < 4; ++kb) {
    v16h a = load_frag_g(wg16, u0 + ln, kb * 32, laneHi);
    v16h b = load_frag_g(h16,  i0 + ln, kb * 32, laneHi);
    acc = __builtin_amdgcn_wmma_f32_16x16x32_f16(false, a, false, b,
                                                 (short)0, acc, false, false);
  }
  // C layout: lanes 0-15 N=lane, VGPR v -> M=v; lanes 16-31 -> M=v+8.
  float* out = pred + (size_t)(u0 + 8 * laneHi) * nI + i0 + ln;
#pragma unroll
  for (int v = 0; v < 8; ++v) out[(size_t)v * nI] = acc[v];
}

// ---------------------------------------------------------------------------
// K4: MLP tower. One wave per strip of TPW consecutive 16-pair tiles (fixed
// user u). W2 fragments (128 VGPRs), A[u], and b2/w_mlp scalars are loaded
// once per strip and amortized over TPW*16 WMMAs. Per tile:
//   h1  = relu(A[u] + B[i])            (16 x 128, built in LDS as f16)
//   h2  = h1 @ W2^T                    (16 WMMAs: 4 N-tiles x 4 K-steps)
//   out = sum_n relu(h2 + b2) * w_mlp  (register relu + shfl_xor row-sum)
//   pred[u, i] += out                  (GMF part already there from K3)
// ---------------------------------------------------------------------------
template <int NI_C, int TPW>
__global__ void k_mlp_wmma(const float* __restrict__ A_mlp,
                           const float* __restrict__ B_mlp,
                           const _Float16* __restrict__ w216,
                           const float* __restrict__ b2,
                           const float* __restrict__ w_mlp,
                           float* __restrict__ pred, int nI_rt, int nStrips) {
  const int nI = NI_C ? NI_C : nI_rt;
  __shared__ _Float16 lds_h1[8][16 * EMB];
  int waveInBlk = threadIdx.x >> 5;
  int lane      = threadIdx.x & 31;
  int laneHi    = lane >> 4;
  int ln        = lane & 15;
  int strip     = blockIdx.x * 8 + waveInBlk;
  if (strip >= nStrips) return;  // wave-uniform guard
  int tilesPerRow  = nI >> 4;
  int stripsPerRow = tilesPerRow / TPW;
  int u     = strip / stripsPerRow;
  int tile0 = (strip % stripsPerRow) * TPW;

  // Loop-invariant: all W2 fragments  B(k,n) = W2[n][k], rows indexed by n.
  v16h bw[4][4];
#pragma unroll
  for (int n = 0; n < 4; ++n)
#pragma unroll
    for (int k = 0; k < 4; ++k)
      bw[n][k] = load_frag_g(w216, n * 16 + ln, k * 32, laneHi);

  // Loop-invariant: per-user layer-1 partial (includes b1) + head scalars.
  float4 va = ((const float4*)(A_mlp + (size_t)u * EMB))[lane];
  float b2v[4], wmv[4];
#pragma unroll
  for (int n = 0; n < 4; ++n) {
    b2v[n] = b2[n * 16 + ln];
    wmv[n] = w_mlp[n * 16 + ln];
  }

  _Float16* tb = &lds_h1[waveInBlk][0];
  for (int t = 0; t < TPW; ++t) {
    int i0 = (tile0 + t) << 4;
    const float* Bbase = B_mlp + (size_t)i0 * EMB;

    // Prefetch next tile's 8KB B-block (global_prefetch_b8, counter-free).
    if (t + 1 < TPW) {
      const char* nxt = (const char*)(Bbase + 16 * EMB);
      __builtin_prefetch(nxt + lane * 256, 0, 0);
      __builtin_prefetch(nxt + lane * 256 + 128, 0, 0);
    }

    // Build h1 tile in LDS (wave-private, DS ops are in-order per wave).
#pragma unroll
    for (int p = 0; p < 16; ++p) {
      float4 bv = ((const float4*)(Bbase + (size_t)p * EMB))[lane];
      v4h hs;
      hs[0] = (_Float16)fmaxf(va.x + bv.x, 0.0f);
      hs[1] = (_Float16)fmaxf(va.y + bv.y, 0.0f);
      hs[2] = (_Float16)fmaxf(va.z + bv.z, 0.0f);
      hs[3] = (_Float16)fmaxf(va.w + bv.w, 0.0f);
      *(v4h*)&tb[p * EMB + lane * 4] = hs;
    }

    float rowsum[8];
#pragma unroll
    for (int v = 0; v < 8; ++v) rowsum[v] = 0.0f;

#pragma unroll
    for (int n = 0; n < 4; ++n) {
      v8f acc = {};
#pragma unroll
      for (int k = 0; k < 4; ++k) {
        const _Float16* p = &tb[ln * EMB + k * 32 + (laneHi ? 8 : 0)];
        v8h lo = *(const v8h*)p;
        v8h hi = *(const v8h*)(p + 16);
        v16h a;
#pragma unroll
        for (int q = 0; q < 8; ++q) { a[q] = lo[q]; a[q + 8] = hi[q]; }
        acc = __builtin_amdgcn_wmma_f32_16x16x32_f16(false, a, false, bw[n][k],
                                                     (short)0, acc, false,
                                                     false);
      }
#pragma unroll
      for (int v = 0; v < 8; ++v) {
        float x = acc[v] + b2v[n];
        x = x > 0.0f ? x : 0.0f;
        rowsum[v] += x * wmv[n];
      }
    }

    // Row-sum across the 16 lanes of each half (masks stay within the half).
#pragma unroll
    for (int m = 8; m >= 1; m >>= 1)
#pragma unroll
      for (int v = 0; v < 8; ++v) rowsum[v] += __shfl_xor(rowsum[v], m, 32);

    if (ln == 0) {
      float* out = pred + (size_t)u * nI + i0 + 8 * laneHi;
#pragma unroll
      for (int v = 0; v < 8; ++v) out[v] += rowsum[v];
    }
  }
}

// ---------------------------------------------------------------------------
extern "C" void kernel_launch(void* const* d_in, const int* in_sizes, int n_in,
                              void* d_out, int out_size, void* d_ws,
                              size_t ws_size, hipStream_t stream) {
  const int*   uidx     = (const int*)d_in[0];
  // d_in[1] = x (unused by the model)
  const int*   iidx     = (const int*)d_in[2];
  const float* user_emb = (const float*)d_in[3];
  const float* item_emb = (const float*)d_in[4];
  const float* W1       = (const float*)d_in[5];
  const float* b1       = (const float*)d_in[6];
  const float* W2       = (const float*)d_in[7];
  const float* b2       = (const float*)d_in[8];
  const float* w_mlp    = (const float*)d_in[9];
  const float* w_gmf    = (const float*)d_in[10];

  const int nI = in_sizes[2];  // 2048

  float* pred  = (float*)d_out;                       // [NU, nI]
  float* w_out = pred + (size_t)NU * nI;              // [NU, EMB]
  float* h_out = w_out + (size_t)NU * EMB;            // [nI, EMB]

  char* ws = (char*)d_ws;
  _Float16* h16  = (_Float16*)ws; ws += (size_t)nI * EMB * sizeof(_Float16);
  _Float16* wg16 = (_Float16*)ws; ws += (size_t)NU * EMB * sizeof(_Float16);
  _Float16* w216 = (_Float16*)ws; ws += (size_t)64 * EMB * sizeof(_Float16);
  float* A_mlp   = (float*)ws;    ws += (size_t)NU * EMB * sizeof(float);
  float* B_mlp   = (float*)ws;

  int totalGather = nI * EMB + NU * EMB + 64 * EMB;
  k_prep_gather<<<(totalGather + 255) / 256, 256, 0, stream>>>(
      uidx, iidx, user_emb, item_emb, W2, w_gmf, w_out, h_out, h16, wg16, w216,
      nI);

  k_prep_mlp<<<NU + nI, EMB, 0, stream>>>(w_out, h_out, W1, b1, A_mlp, B_mlp);

  int gmfTiles = (NU / 16) * (nI / 16);  // 16x16 tiles of pred
  int gmfBlocks = (gmfTiles + 7) / 8;

  if (nI == 2048) {
    k_gmf_wmma<2048><<<gmfBlocks, 256, 0, stream>>>(wg16, h16, pred, nI,
                                                    gmfTiles);
    const int TPW = 8;
    int strips = NU * ((2048 / 16) / TPW);           // 4096
    k_mlp_wmma<2048, TPW><<<(strips + 7) / 8, 256, 0, stream>>>(
        A_mlp, B_mlp, w216, b2, w_mlp, pred, nI, strips);
  } else {
    k_gmf_wmma<0><<<gmfBlocks, 256, 0, stream>>>(wg16, h16, pred, nI,
                                                 gmfTiles);
    int strips = NU * (nI / 16);                     // TPW = 1 fallback
    k_mlp_wmma<0, 1><<<(strips + 7) / 8, 256, 0, stream>>>(
        A_mlp, B_mlp, w216, b2, w_mlp, pred, nI, strips);
  }
}
